// SelfAttentionHead_58394375356490
// MI455X (gfx1250) — matrix operand order repaired and checked by reading, not compiled, not run on hardware
//
#include <hip/hip_runtime.h>
#include <hip/hip_bf16.h>
#include <stdint.h>

// ---------------------------------------------------------------------------
// Self-attention head, CDNA5 / gfx1250, bf16 WMMA flash-attention.
// B=8 T=2048 D=1024 H=128, fp32 in/out, bf16 matmul with fp32 accumulate.
// Attention: block-level flash attention, 128 queries/block, 32-key steps,
// K/V staged in LDS via async global->LDS copies (double buffered).
// ---------------------------------------------------------------------------

typedef __attribute__((ext_vector_type(16))) __bf16 v16bf;
typedef __attribute__((ext_vector_type(8)))  float  v8f;

constexpr int Bb = 8, Tt = 2048, Dd = 1024, Hh = 128;
constexpr int Mm = Bb * Tt;              // 16384 rows

#if defined(__has_builtin)
#if __has_builtin(__builtin_amdgcn_global_load_async_to_lds_b128)
#define HAVE_ASYNC_LDS 1
#endif
#endif
#ifndef HAVE_ASYNC_LDS
#define HAVE_ASYNC_LDS 0
#endif

__device__ __forceinline__ uint16_t f2bf(float f) {
    uint32_t u = __builtin_bit_cast(uint32_t, f);
    u += 0x7fffu + ((u >> 16) & 1u);     // round-to-nearest-even
    return (uint16_t)(u >> 16);
}

union F16x16 { v16bf v; uint32_t u[8]; };

// Pointer types matching the async-to-LDS builtin signature:
// param0: addrspace(1) pointer to <4 x i32>, param1: addrspace(3) pointer.
typedef int v4i_vec __attribute__((vector_size(16)));
typedef __attribute__((address_space(1))) v4i_vec* gptr_v4i;
typedef __attribute__((address_space(3))) v4i_vec* lptr_v4i;

// 16-byte global -> LDS copy; async (ASYNCcnt) when the builtin exists.
__device__ __forceinline__ void copy16_g2l(const uint16_t* g, uint16_t* l) {
#if HAVE_ASYNC_LDS
    __builtin_amdgcn_global_load_async_to_lds_b128((gptr_v4i)g, (lptr_v4i)l, 0, 0);
#else
    *(uint4*)l = *(const uint4*)g;
#endif
}

__device__ __forceinline__ void wait_async_tiles(bool keep4) {
#if HAVE_ASYNC_LDS
    if (keep4) asm volatile("s_wait_asynccnt 4" ::: "memory");
    else       asm volatile("s_wait_asynccnt 0" ::: "memory");
#else
    (void)keep4;
#endif
}

// ---------------------------------------------------------------------------
// Elementwise fp32 -> bf16
__global__ __launch_bounds__(256) void cvt_f32_bf16(const float* __restrict__ in,
                                                    uint16_t* __restrict__ out, int n) {
    int i = blockIdx.x * 256 + threadIdx.x;
    if (i < n) out[i] = f2bf(in[i]);
}

// W [D,H] fp32 -> Wt [H,D] bf16 (so K-pairs of the B-fragment are contiguous)
__global__ __launch_bounds__(256) void cvt_w_T(const float* __restrict__ w,
                                               uint16_t* __restrict__ out) {
    int i = blockIdx.x * 256 + threadIdx.x;
    if (i < Dd * Hh) {
        int d = i / Hh, h = i % Hh;
        out[(size_t)h * Dd + d] = f2bf(w[i]);
    }
}

// ---------------------------------------------------------------------------
// Projections: q = x@Wq, k = x@Wk (stored [M,H] bf16), v = x@Wv (stored [B,H,T] bf16)
// One wave per 16x16 output tile; K-loop over D in steps of 32.
__global__ __launch_bounds__(256) void proj_kernel(const uint16_t* __restrict__ xb,
                                                   const uint16_t* __restrict__ wtq,
                                                   const uint16_t* __restrict__ wtk,
                                                   const uint16_t* __restrict__ wtv,
                                                   uint16_t* __restrict__ q_ws,
                                                   uint16_t* __restrict__ k_ws,
                                                   uint16_t* __restrict__ v_ws) {
    int lane = threadIdx.x & 31;
    int l15  = lane & 15, half = lane >> 4;
    int wv   = blockIdx.x * 8 + (threadIdx.x >> 5);
    int mt    = wv / 24;
    int rem   = wv % 24;
    int which = rem >> 3;                // 0:q 1:k 2:v
    int nt    = rem & 7;

    const uint16_t* wt = (which == 0) ? wtq : (which == 1) ? wtk : wtv;

    int m_a = mt * 16 + l15;             // A-fragment row (both lane halves cover M=0..15)
    int n_b = nt * 16 + l15;             // B-fragment column

    v8f c = {};
    for (int k0 = 0; k0 < Dd; k0 += 32) {
        F16x16 a, b;
#pragma unroll
        for (int v = 0; v < 8; ++v) {
            int ka = k0 + ((v < 4) ? 0 : 16) + half * 8 + (v & 3) * 2;   // A 16x32 layout
            a.u[v] = *(const uint32_t*)(xb + (size_t)m_a * Dd + ka);
            int kb = k0 + ((v < 4) ? 0 : 16) + half * 8 + (v & 3) * 2;   // B 32x16 layout
            b.u[v] = *(const uint32_t*)(wt + (size_t)n_b * Dd + kb);
        }
        c = __builtin_amdgcn_wmma_f32_16x16x32_bf16(false, a.v, false, b.v,
                                                    (short)0, c, false, false);
    }

#pragma unroll
    for (int r = 0; r < 8; ++r) {
        int m = mt * 16 + half * 8 + r;  // C layout: lanes 0-15 rows 0-7, lanes 16-31 rows 8-15
        uint16_t val = f2bf(c[r]);
        if (which == 0)      q_ws[(size_t)m * Hh + nt * 16 + l15] = val;
        else if (which == 1) k_ws[(size_t)m * Hh + nt * 16 + l15] = val;
        else {
            int bb = m / Tt, t = m % Tt; // v transposed: [B,H,T]
            v_ws[((size_t)bb * Hh + nt * 16 + l15) * Tt + t] = val;
        }
    }
}

// ---------------------------------------------------------------------------
// Block-level flash attention:
//   block = 8 waves = 128 query rows (one batch), streaming 32 keys/step.
//   K/V 32-key tiles staged in LDS by async global->LDS copies, double buffered.
__global__ __launch_bounds__(256) void attn_kernel(const uint16_t* __restrict__ q_ws,
                                                   const uint16_t* __restrict__ k_ws,
                                                   const uint16_t* __restrict__ v_ws,
                                                   float* __restrict__ out) {
    // Padded rows: kT stride 136 halves (272B), vT stride 40 halves (80B) -> no bank camping.
    __shared__ uint16_t kT[2][32][136];      // [buf][key][h]
    __shared__ uint16_t vT[2][128][40];      // [buf][h][key]
    __shared__ uint16_t pbuf[8][16 * 32];    // per-wave probs tile (C-layout -> A-layout)

    int tid   = threadIdx.x;
    int lane  = tid & 31;
    int wslot = tid >> 5;
    int l15   = lane & 15, half = lane >> 4;

    int b   = blockIdx.x >> 4;               // 16 query-blocks per batch
    int qb  = blockIdx.x & 15;
    int tq0 = qb * 128;
    int tq  = tq0 + wslot * 16;              // this wave's 16 rows
    size_t bT = (size_t)b * Tt;
    size_t bH = (size_t)b * Hh;

    // ---- Q fragments for all of H (4 K-steps of 32), kept in registers ----
    F16x16 qf[4];
#pragma unroll
    for (int ks = 0; ks < 4; ++ks)
#pragma unroll
        for (int v = 0; v < 8; ++v) {
            int d = ks * 32 + ((v < 4) ? 0 : 16) + half * 8 + (v & 3) * 2;
            qf[ks].u[v] = *(const uint32_t*)(q_ws + (bT + tq + l15) * Hh + d);
        }

    v8f acc[8];
#pragma unroll
    for (int h = 0; h < 8; ++h) acc[h] = (v8f){0, 0, 0, 0, 0, 0, 0, 0};
    float mi[8], li[8];
#pragma unroll
    for (int r = 0; r < 8; ++r) { mi[r] = -3.0e38f; li[r] = 0.0f; }

    const float scale = 0.08838834764831845f;   // 1/sqrt(128)
    const float LOG2E = 1.4426950408889634f;

    int nsteps = (tq0 + 159) >> 5;              // cover keys 0..tq0+127

    // ---- tile fetch: 512 16B segments (2 per thread for K, 2 for V) ----
    auto issue_tiles = [&](int step) {
        int buf = step & 1, kb = step << 5;
#pragma unroll
        for (int j = 0; j < 2; ++j) {
            int idx = j * 256 + tid;
            int row = idx >> 4, seg = idx & 15;                 // K: 32 rows x 16 segs
            copy16_g2l(k_ws + (bT + kb + row) * Hh + seg * 8, &kT[buf][row][seg * 8]);
            int hh = idx >> 2, sg = idx & 3;                    // V: 128 rows x 4 segs
            copy16_g2l(v_ws + (bH + hh) * Tt + kb + sg * 8, &vT[buf][hh][sg * 8]);
        }
    };

    issue_tiles(0);

    for (int s = 0; s < nsteps; ++s) {
        int kb = s << 5;
        int buf = s & 1;

        __syncthreads();                        // everyone done reading buf^1 (step s-1)
        bool pf = (s + 1) < nsteps;
        if (pf) issue_tiles(s + 1);             // prefetch into buf^1
        wait_async_tiles(pf);                   // this wave's tiles(s) complete
        __syncthreads();                        // all waves' tiles(s) visible

        if (kb > tq + 15) continue;             // fully masked for this wave's rows

        // ---- scores: two 16x16 tiles (keys kb..kb+15, kb+16..kb+31) over H ----
        v8f c0 = {}, c1 = {};
#pragma unroll
        for (int ks = 0; ks < 4; ++ks) {
            F16x16 kf0, kf1;
            const uint16_t* kp0 = &kT[buf][l15][ks * 32 + half * 8];
            const uint16_t* kp1 = &kT[buf][16 + l15][ks * 32 + half * 8];
#pragma unroll
            for (int v = 0; v < 8; ++v) {
                kf0.u[v] = *(const uint32_t*)(kp0 + ((v < 4) ? 0 : 16) + (v & 3) * 2);
                kf1.u[v] = *(const uint32_t*)(kp1 + ((v < 4) ? 0 : 16) + (v & 3) * 2);
            }
            c0 = __builtin_amdgcn_wmma_f32_16x16x32_bf16(false, qf[ks].v, false, kf0.v,
                                                         (short)0, c0, false, false);
            c1 = __builtin_amdgcn_wmma_f32_16x16x32_bf16(false, qf[ks].v, false, kf1.v,
                                                         (short)0, c1, false, false);
        }

        // ---- causal mask + online softmax (rows live in C layout) ----
        int key0 = kb + l15, key1 = kb + 16 + l15;
        float p0[8], p1[8], alpha[8];
#pragma unroll
        for (int r = 0; r < 8; ++r) {
            int row = tq + half * 8 + r;
            float s0 = (key0 <= row) ? c0[r] * scale : -3.0e38f;
            float s1 = (key1 <= row) ? c1[r] * scale : -3.0e38f;
            float lm = fmaxf(s0, s1);
#pragma unroll
            for (int msk = 8; msk >= 1; msk >>= 1)
                lm = fmaxf(lm, __shfl_xor(lm, msk, 32));   // stays within 16-lane half
            float nm = fmaxf(mi[r], lm);
            alpha[r] = exp2f((mi[r] - nm) * LOG2E);
            p0[r] = exp2f((s0 - nm) * LOG2E);
            p1[r] = exp2f((s1 - nm) * LOG2E);
            float rs = p0[r] + p1[r];
#pragma unroll
            for (int msk = 8; msk >= 1; msk >>= 1)
                rs += __shfl_xor(rs, msk, 32);
            li[r] = li[r] * alpha[r] + rs;
            mi[r] = nm;
        }
#pragma unroll
        for (int h = 0; h < 8; ++h)
#pragma unroll
            for (int r = 0; r < 8; ++r) acc[h][r] *= alpha[r];

        // ---- probs: C layout -> per-wave LDS tile -> A-fragment ----
#pragma unroll
        for (int r = 0; r < 8; ++r) {
            int row = half * 8 + r;
            pbuf[wslot][row * 32 + l15]      = f2bf(p0[r]);
            pbuf[wslot][row * 32 + 16 + l15] = f2bf(p1[r]);
        }
        asm volatile("s_wait_dscnt 0" ::: "memory");
        F16x16 pa;
#pragma unroll
        for (int v = 0; v < 8; ++v) {
            int kk = ((v < 4) ? 0 : 16) + half * 8 + (v & 3) * 2;
            pa.u[v] = *(const uint32_t*)(&pbuf[wslot][l15 * 32 + kk]);
        }

        // ---- PV: acc[16 x 128] += P[16x32] @ V[32x128] (V tile in LDS, [h][key]) ----
#pragma unroll
        for (int h = 0; h < 8; ++h) {
            F16x16 vf;
            const uint16_t* vp = &vT[buf][h * 16 + l15][half * 8];
#pragma unroll
            for (int v = 0; v < 8; ++v)
                vf.u[v] = *(const uint32_t*)(vp + ((v < 4) ? 0 : 16) + (v & 3) * 2);
            acc[h] = __builtin_amdgcn_wmma_f32_16x16x32_bf16(false, pa.v, false, vf.v,
                                                             (short)0, acc[h], false, false);
        }
    }

    // ---- epilogue: normalize and store fp32 ----
#pragma unroll
    for (int r = 0; r < 8; ++r) {
        float inv = 1.0f / li[r];
#pragma unroll
        for (int h = 0; h < 8; ++h)
            out[(bT + tq + half * 8 + r) * Hh + h * 16 + l15] = acc[h][r] * inv;
    }
}

// ---------------------------------------------------------------------------
extern "C" void kernel_launch(void* const* d_in, const int* in_sizes, int n_in,
                              void* d_out, int out_size, void* d_ws, size_t ws_size,
                              hipStream_t stream) {
    const float* x  = (const float*)d_in[0];
    const float* Wq = (const float*)d_in[1];
    const float* Wk = (const float*)d_in[2];
    const float* Wv = (const float*)d_in[3];
    float* out = (float*)d_out;

    // workspace layout (bf16 elements): x_bf | WqT | WkT | WvT | q | k | vT
    uint16_t* ws   = (uint16_t*)d_ws;
    uint16_t* xb   = ws;
    uint16_t* wtq  = xb  + (size_t)Mm * Dd;
    uint16_t* wtk  = wtq + (size_t)Hh * Dd;
    uint16_t* wtv  = wtk + (size_t)Hh * Dd;
    uint16_t* q_ws = wtv + (size_t)Hh * Dd;
    uint16_t* k_ws = q_ws + (size_t)Mm * Hh;
    uint16_t* v_ws = k_ws + (size_t)Mm * Hh;

    int nx = Mm * Dd;
    cvt_f32_bf16<<<nx / 256, 256, 0, stream>>>(x, xb, nx);
    cvt_w_T<<<(Dd * Hh) / 256, 256, 0, stream>>>(Wq, wtq);
    cvt_w_T<<<(Dd * Hh) / 256, 256, 0, stream>>>(Wk, wtk);
    cvt_w_T<<<(Dd * Hh) / 256, 256, 0, stream>>>(Wv, wtv);

    // 1024 m-tiles * 8 n-tiles * 3 matrices = 24576 wave-tiles, 8 waves/block
    proj_kernel<<<3072, 256, 0, stream>>>(xb, wtq, wtk, wtv, q_ws, k_ws, v_ws);

    // 8 batches * 16 query-blocks (128 rows each) = 128 blocks of 8 waves
    attn_kernel<<<128, 256, 0, stream>>>(q_ws, k_ws, v_ws, out);
}
